// Transformer_61040075210877
// MI455X (gfx1250) — matrix operand order repaired and checked by reading, not compiled, not run on hardware
//
#include <hip/hip_runtime.h>
#include <hip/hip_bf16.h>
#include <hip/hip_fp16.h>
#include <math.h>

// ---------------------------------------------------------------------------
// CDNA5 (gfx1250) transformer forward.
//  - All matmuls: V_WMMA_F32_16X16X32_F16 (f16 in, f32 accumulate)
//  - Projection GEMMs: A-tile staged to LDS via Tensor Data Mover
//    (TENSOR_LOAD_TO_LDS + S_WAIT_TENSORCNT), shared by 8 waves;
//    B fragments are contiguous b128 loads from f16-preconverted weights.
// ---------------------------------------------------------------------------

#define DEPTH 12
#define HEADS 16
#define DIM   1024
#define MLP_  4096
#define JTOK  17
#define P0    256
#define BATCH 8
#define N0    273            // J + P0
#define NP    288            // padded token rows (mult of 32)
#define HD    64
#define SCALE_Q 0.03125f     // DIM^-0.5
#define KC_STAGE 256         // K-chunk staged into LDS per iteration

#define CEILDIV(a,b) (((a)+(b)-1)/(b))

typedef _Float16 v8h  __attribute__((ext_vector_type(8)));
typedef _Float16 v16h __attribute__((ext_vector_type(16)));
typedef float    v8f  __attribute__((ext_vector_type(8)));
typedef unsigned int u32x4_t __attribute__((ext_vector_type(4)));
typedef int          i32x4_t __attribute__((ext_vector_type(4)));
typedef int          i32x8_t __attribute__((ext_vector_type(8)));

// ---- TDM availability -------------------------------------------------------
#if defined(__has_builtin)
#if __has_builtin(__builtin_amdgcn_tensor_load_to_lds)
#define HAVE_TDM 1
#endif
#endif
#ifndef HAVE_TDM
#define HAVE_TDM 0
#endif

#if HAVE_TDM
#if __has_builtin(__builtin_amdgcn_s_wait_tensorcnt)
#define TDM_WAIT0() __builtin_amdgcn_s_wait_tensorcnt(0)
#else
#define TDM_WAIT0() asm volatile("s_wait_tensorcnt 0" ::: "memory")
#endif

// 2D tile load: 16 rows x KC f16 elements, row stride K elements, into LDS.
// Descriptor per CDNA5 ISA.
__device__ __forceinline__ void tdm_load_a_tile(const _Float16* gsrc,
                                                _Float16* lds, int K, int KC) {
  const unsigned long long ga = (unsigned long long)(uintptr_t)gsrc;
  const unsigned ldsOff = (unsigned)(uintptr_t)lds;  // addr[31:0] = LDS offset

  u32x4_t g0 = {};
  g0[0] = 1u;                                        // count=1, user mode
  g0[1] = ldsOff;                                    // lds_addr (bytes)
  g0[2] = (unsigned)(ga & 0xFFFFFFFFu);              // global_addr[31:0]
  g0[3] = (unsigned)((ga >> 32) & 0x01FFFFFFu)       // global_addr[56:32]
          | 0x80000000u;                             // type = 2 ("image")

  i32x8_t g1 = {};
  g1[0] = (int)(1u << 16);                           // data_size=1 -> 2 bytes
  g1[1] = (int)((unsigned)KC << 16);                 // tensor_dim0[15:0]
  g1[2] = (int)(((unsigned)KC >> 16) | (16u << 16)); // dim0[31:16] | dim1[15:0]
  g1[3] = (int)((unsigned)KC << 16);                 // dim1[31:16]=0 | tile_dim0
  g1[4] = 16;                                        // tile_dim1=16, tile_dim2=0
  g1[5] = (int)(unsigned)K;                          // tensor_dim0_stride[31:0]
  g1[6] = 0;                                         // stride[47:32] | dim1_stride lo
  g1[7] = 0;

  i32x4_t z4 = {};
#if __has_include(<hip/amd_detail/amd_gfx1250_TDM.h>)
  i32x8_t z8 = {};
  __builtin_amdgcn_tensor_load_to_lds(g0, g1, z4, z4, z8, 0);
#else
  __builtin_amdgcn_tensor_load_to_lds(g0, g1, z4, z4, 0);
#endif
}
#endif  // HAVE_TDM

__device__ __forceinline__ v16h make_v16(v8h lo, v8h hi) {
  v16h r;
#pragma unroll
  for (int i = 0; i < 8; ++i) { r[i] = lo[i]; r[i + 8] = hi[i]; }
  return r;
}

// A-matrix fragment (16x32 f16): lane holds row (lane&15); K = half*16 + e.
__device__ __forceinline__ v16h load_a_frag(const _Float16* rowPtrAtK0, int half) {
  v8h c0 = *(const v8h*)(rowPtrAtK0 + half * 16);
  v8h c1 = *(const v8h*)(rowPtrAtK0 + half * 16 + 8);
  return make_v16(c0, c1);
}

// B-matrix fragment (32x16): lane holds column (lane&15); K = half*16 + e.
__device__ __forceinline__ v16h load_b_contig_f16(const _Float16* p) {
  v8h c0 = *(const v8h*)p;
  v8h c1 = *(const v8h*)(p + 8);
  return make_v16(c0, c1);
}

__device__ __forceinline__ v8f wmma_f16(v16h a, v16h b, v8f c) {
  return __builtin_amdgcn_wmma_f32_16x16x32_f16(false, a, false, b, (short)0, c,
                                                false, false);
}

// ---------------------------------------------------------------------------
// Weight conversion f32 -> f16 (counts are multiples of 4)
// ---------------------------------------------------------------------------
__global__ __launch_bounds__(256) void k_cvt_f16(const float* __restrict__ src,
                                                 _Float16* __restrict__ dst,
                                                 size_t count) {
  const size_t i = ((size_t)blockIdx.x * 256 + threadIdx.x) * 4;
  if (i >= count) return;
#pragma unroll
  for (int j = 0; j < 4; ++j) dst[i + j] = (_Float16)src[i + j];
}

// ---------------------------------------------------------------------------
// GEMM: out[b][m][n] = sum_k A16[b][m][k] * W16[n][k]  (+bias,+gelu,+resid)
// A-tile (16 x KC_STAGE) staged to LDS by the TDM, shared across 8 waves.
// grid: x = N/512, y = Mtiles, z = BATCH; block 256 = 8 waves x 4 N-tiles
// ---------------------------------------------------------------------------
template <bool BIAS, bool GELU, bool RESID, bool OUT_F32, bool OUT_F16>
__global__ __launch_bounds__(256) void k_gemm4(
    const _Float16* __restrict__ A, const _Float16* __restrict__ W16,
    const float* __restrict__ bias, const float* __restrict__ resid,
    float* __restrict__ outF, _Float16* __restrict__ outH, int K, int N) {
  __shared__ __align__(16) _Float16 aT[16 * KC_STAGE];

  const int wave = threadIdx.x >> 5, lane = threadIdx.x & 31;
  const int half = lane >> 4, low = lane & 15;
  const int nt0 = (blockIdx.x * 8 + wave) * 4;   // first of 4 N-tiles
  const int mt = blockIdx.y;
  const int b  = blockIdx.z;

  const _Float16* Abase = A + (size_t)(b * NP + mt * 16) * K;  // tile row 0
  const _Float16* W0    = W16 + (size_t)(nt0 * 16 + low) * K + half * 16;
  const size_t wstep = (size_t)16 * K;

  v8f acc0 = {}, acc1 = {}, acc2 = {}, acc3 = {};
  for (int kc = 0; kc < K; kc += KC_STAGE) {
#if HAVE_TDM
    if (threadIdx.x < 32) {           // wave 0 issues the DMA and waits on it
      tdm_load_a_tile(Abase + kc, aT, K, KC_STAGE);
      TDM_WAIT0();
    }
#else
    for (int i = threadIdx.x; i < 16 * KC_STAGE / 8; i += 256) {
      const int rr = (i * 8) / KC_STAGE, cc = (i * 8) % KC_STAGE;
      *(v8h*)(aT + rr * KC_STAGE + cc) =
          *(const v8h*)(Abase + (size_t)rr * K + kc + cc);
    }
#endif
    __syncthreads();

    const _Float16* ar = aT + low * KC_STAGE;
    for (int k0 = 0; k0 < KC_STAGE; k0 += 32) {
      const v16h a = load_a_frag(ar + k0, half);   // ds_load_b128 x2
      acc0 = wmma_f16(a, load_b_contig_f16(W0 + kc + k0), acc0);
      acc1 = wmma_f16(a, load_b_contig_f16(W0 + wstep + kc + k0), acc1);
      acc2 = wmma_f16(a, load_b_contig_f16(W0 + 2 * wstep + kc + k0), acc2);
      acc3 = wmma_f16(a, load_b_contig_f16(W0 + 3 * wstep + kc + k0), acc3);
    }
    __syncthreads();
  }

#pragma unroll
  for (int t = 0; t < 4; ++t) {
    const v8f acc = (t == 0) ? acc0 : (t == 1) ? acc1 : (t == 2) ? acc2 : acc3;
    const int n0 = (nt0 + t) * 16 + low;
#pragma unroll
    for (int r = 0; r < 8; ++r) {
      const int m = mt * 16 + r + 8 * half;
      const size_t o = (size_t)(b * NP + m) * N + n0;
      float v = acc[r];
      if (BIAS)  v += bias[n0];
      if (GELU)  v = 0.5f * v * (1.0f + erff(v * 0.70710678118f));
      if (RESID) v += resid[o];
      if (OUT_F32) outF[o] = v;
      if (OUT_F16) outH[o] = (_Float16)v;
    }
  }
}

// ---------------------------------------------------------------------------
// Attention scores: sc[bh][i][j] = sum_d q[bh][i][d] * k[bh][j][d]   (K = 64)
// grid: x = ceil(tiles/8), y = tiles, z = BATCH*HEADS
// ---------------------------------------------------------------------------
__global__ __launch_bounds__(256) void k_scores(const _Float16* __restrict__ q,
                                                const _Float16* __restrict__ kk,
                                                float* __restrict__ sc, int n) {
  const int wave = threadIdx.x >> 5, lane = threadIdx.x & 31;
  const int half = lane >> 4, low = lane & 15;
  const int tiles = CEILDIV(n, 16);
  int nt = blockIdx.x * 8 + wave;
  if (nt >= tiles) nt = tiles - 1;  // duplicate work, identical writes
  const int mt = blockIdx.y;
  const int bh = blockIdx.z;

  const _Float16* qrow = q  + (size_t)(bh * NP + mt * 16 + low) * HD;
  const _Float16* krow = kk + (size_t)(bh * NP + nt * 16 + low) * HD + half * 16;

  v8f acc = {};
#pragma unroll
  for (int k0 = 0; k0 < HD; k0 += 32) {
    v16h af = load_a_frag(qrow + k0, half);
    v16h bf = load_b_contig_f16(krow + k0);
    acc = wmma_f16(af, bf, acc);
  }

#pragma unroll
  for (int r = 0; r < 8; ++r) {
    const int m = mt * 16 + r + 8 * half;
    sc[(size_t)bh * NP * NP + (size_t)m * NP + nt * 16 + low] = acc[r];
  }
}

// ---------------------------------------------------------------------------
// attn @ V: o[b][m][h*64+c] = sum_j attn16[bh][m][j] * vT[bh][c][j]
// vT is transposed [BH][HD][NP] so B-fragments are contiguous b128 loads.
// grid: x = 1, y = ceil(Mtiles/2), z = BATCH*HEADS; 8 waves = 2 Mt x 4 Nt
// ---------------------------------------------------------------------------
__global__ __launch_bounds__(256) void k_attnv(const _Float16* __restrict__ attn,
                                               const _Float16* __restrict__ vT,
                                               _Float16* __restrict__ o, int n) {
  const int wave = threadIdx.x >> 5, lane = threadIdx.x & 31;
  const int half = lane >> 4, low = lane & 15;
  const int ntile = wave & 3;
  const int mtiles = CEILDIV(n, 16);
  int mt = blockIdx.y * 2 + (wave >> 2);
  if (mt >= mtiles) mt = mtiles - 1;
  const int bh = blockIdx.z;
  const int b = bh >> 4, h = bh & 15;
  const int Kp = (n + 31) & ~31;

  const _Float16* arow = attn + (size_t)bh * NP * NP + (size_t)(mt * 16 + low) * NP;
  const _Float16* vrow = vT + (size_t)(bh * HD + ntile * 16 + low) * NP + half * 16;

  v8f acc = {};
  for (int k0 = 0; k0 < Kp; k0 += 32) {
    v16h af = load_a_frag(arow + k0, half);
    v16h bf = load_b_contig_f16(vrow + k0);
    acc = wmma_f16(af, bf, acc);
  }

#pragma unroll
  for (int r = 0; r < 8; ++r) {
    const int m = mt * 16 + r + 8 * half;
    o[(size_t)(b * NP + m) * DIM + h * HD + ntile * 16 + low] = (_Float16)acc[r];
  }
}

// ---------------------------------------------------------------------------
// LayerNorm -> f16.  grid: (n, BATCH), block 256
// ---------------------------------------------------------------------------
__global__ __launch_bounds__(256) void k_ln(const float* __restrict__ x,
                                            const float* __restrict__ g,
                                            const float* __restrict__ bt,
                                            _Float16* __restrict__ out) {
  const int m = blockIdx.x, b = blockIdx.y, tid = threadIdx.x;
  const float* row = x + (size_t)(b * NP + m) * DIM;
  __shared__ float sh[256], sh2[256];
  float s = 0.f, s2 = 0.f;
  for (int c = tid; c < DIM; c += 256) { float v = row[c]; s += v; s2 += v * v; }
  sh[tid] = s; sh2[tid] = s2; __syncthreads();
  for (int st = 128; st > 0; st >>= 1) {
    if (tid < st) { sh[tid] += sh[tid + st]; sh2[tid] += sh2[tid + st]; }
    __syncthreads();
  }
  const float mean = sh[0] * (1.0f / DIM);
  const float var  = sh2[0] * (1.0f / DIM) - mean * mean;
  const float inv  = rsqrtf(var + 1e-5f);
  _Float16* orow = out + (size_t)(b * NP + m) * DIM;
  for (int c = tid; c < DIM; c += 256)
    orow[c] = (_Float16)((row[c] - mean) * inv * g[c] + bt[c]);
}

// ---------------------------------------------------------------------------
// RoPE split: qkv f32 -> q(*scale,rope) k(rope) [BH][NP][HD]; v -> vT [BH][HD][NP]
// grid: (n, BATCH), block 256
// ---------------------------------------------------------------------------
__global__ __launch_bounds__(256) void k_rope(const float* __restrict__ qkv,
                                              const float* __restrict__ fc,
                                              _Float16* __restrict__ qF,
                                              _Float16* __restrict__ kF,
                                              _Float16* __restrict__ vT) {
  const int m = blockIdx.x, b = blockIdx.y;
  const float* base = qkv + (size_t)(b * NP + m) * (3 * DIM);
  for (int pp = threadIdx.x; pp < HEADS * (HD / 2); pp += 256) {
    const int h = pp >> 5, t = pp & 31;
    const float c = fc[(m * 32 + t) * 2 + 0];
    const float s = fc[(m * 32 + t) * 2 + 1];
    const int ch = h * HD + 2 * t;
    const float qa = base[ch], qb = base[ch + 1];
    const float ka = base[DIM + ch], kb = base[DIM + ch + 1];
    const float va = base[2 * DIM + ch], vb = base[2 * DIM + ch + 1];
    const int bh = b * HEADS + h;
    const size_t o = (size_t)(bh * NP + m) * HD + 2 * t;
    qF[o]     = (_Float16)((qa * c - qb * s) * SCALE_Q);
    qF[o + 1] = (_Float16)((qa * s + qb * c) * SCALE_Q);
    kF[o]     = (_Float16)(ka * c - kb * s);
    kF[o + 1] = (_Float16)(ka * s + kb * c);
    vT[((size_t)bh * HD + 2 * t) * NP + m]     = (_Float16)va;
    vT[((size_t)bh * HD + 2 * t + 1) * NP + m] = (_Float16)vb;
  }
}

// ---------------------------------------------------------------------------
// Softmax row: writes f32 attn to d_out (tight n x n) + zero-padded f16 copy.
// grid: (n, BATCH*HEADS), block 256
// ---------------------------------------------------------------------------
__global__ __launch_bounds__(256) void k_softmax(const float* __restrict__ sc,
                                                 float* __restrict__ aout,
                                                 _Float16* __restrict__ a16, int n) {
  const int i = blockIdx.x, bh = blockIdx.y, tid = threadIdx.x;
  const float* row = sc + (size_t)bh * NP * NP + (size_t)i * NP;
  __shared__ float sh[256];
  float mx = -3.4e38f;
  for (int j = tid; j < n; j += 256) mx = fmaxf(mx, row[j]);
  sh[tid] = mx; __syncthreads();
  for (int st = 128; st > 0; st >>= 1) {
    if (tid < st) sh[tid] = fmaxf(sh[tid], sh[tid + st]);
    __syncthreads();
  }
  mx = sh[0]; __syncthreads();
  float sm = 0.f;
  for (int j = tid; j < n; j += 256) sm += expf(row[j] - mx);
  sh[tid] = sm; __syncthreads();
  for (int st = 128; st > 0; st >>= 1) {
    if (tid < st) sh[tid] += sh[tid + st];
    __syncthreads();
  }
  const float inv = 1.0f / sh[0];
  float* orow = aout + ((size_t)bh * n + i) * n;
  _Float16* hrow = a16 + (size_t)bh * NP * NP + (size_t)i * NP;
  for (int j = tid; j < n; j += 256) {
    const float e = expf(row[j] - mx) * inv;
    orow[j] = e;
    hrow[j] = (_Float16)e;
  }
  for (int j = n + tid; j < NP; j += 256) hrow[j] = (_Float16)0.0f;
}

// ---------------------------------------------------------------------------
// Pruning: per-token attention score  human[b][p] = mean_h sum_{i<J} attn[...,J+p]
// ---------------------------------------------------------------------------
__global__ __launch_bounds__(256) void k_prune_scores(const float* __restrict__ aout,
                                                      float* __restrict__ human,
                                                      int n) {
  const int P = n - JTOK;
  const int idx = blockIdx.x * 256 + threadIdx.x;
  if (idx >= BATCH * P) return;
  const int b = idx / P, p = idx % P;
  float s = 0.f;
  for (int h = 0; h < HEADS; ++h)
    for (int i = 0; i < JTOK; ++i)
      s += aout[((size_t)(b * HEADS + h) * n + i) * n + JTOK + p];
  human[b * P0 + p] = s * (1.0f / HEADS);
}

// Bitonic top-k over <=256 scores/batch; descending value, ascending index ties.
__global__ __launch_bounds__(256) void k_topk(const float* __restrict__ human,
                                              int* __restrict__ out, int P, int k) {
  __shared__ float sv[256];
  __shared__ int   si[256];
  const int b = blockIdx.x, t = threadIdx.x;
  sv[t] = (t < P) ? human[b * P0 + t] : -3.4e38f;
  si[t] = t;
  __syncthreads();
  for (int size = 2; size <= 256; size <<= 1) {
    for (int stride = size >> 1; stride > 0; stride >>= 1) {
      const int j = t ^ stride;
      if (j > t) {
        const bool desc = ((t & size) == 0);
        const float a = sv[t], bb = sv[j];
        const int ia = si[t], ib = si[j];
        const bool aFirst = (a > bb) || (a == bb && ia < ib);
        if (desc ? !aFirst : aFirst) { sv[t] = bb; sv[j] = a; si[t] = ib; si[j] = ia; }
      }
      __syncthreads();
    }
  }
  if (t < k) out[b * P0 + t] = si[t];
}

__global__ __launch_bounds__(256) void k_gather_x(const float* __restrict__ x,
                                                  float* __restrict__ xn,
                                                  const int* __restrict__ top, int k) {
  const int rows = JTOK + k;
  const size_t tot = (size_t)BATCH * rows * DIM;
  const size_t idx = (size_t)blockIdx.x * 256 + threadIdx.x;
  if (idx >= tot) return;
  const int c = (int)(idx % DIM);
  const size_t r = idx / DIM;
  const int m = (int)(r % rows), b = (int)(r / rows);
  const int src = (m < JTOK) ? m : (JTOK + top[b * P0 + (m - JTOK)]);
  xn[((size_t)(b * NP) + m) * DIM + c] = x[((size_t)(b * NP) + src) * DIM + c];
}

__global__ __launch_bounds__(256) void k_gather_pos(const float* __restrict__ pos,
                                                    float* __restrict__ pn,
                                                    const int* __restrict__ top, int k) {
  const size_t tot = (size_t)BATCH * k * DIM;
  const size_t idx = (size_t)blockIdx.x * 256 + threadIdx.x;
  if (idx >= tot) return;
  const int c = (int)(idx % DIM);
  const size_t r = idx / DIM;
  const int q = (int)(r % k), b = (int)(r / k);
  pn[((size_t)b * P0 + q) * DIM + c] = pos[((size_t)b * P0 + top[b * P0 + q]) * DIM + c];
}

// pos *= la[layer]; x[:, J:J+P] += pos
__global__ __launch_bounds__(256) void k_pos_add(float* __restrict__ pos,
                                                 float* __restrict__ x,
                                                 const float* __restrict__ la,
                                                 int layer, int P) {
  const size_t tot = (size_t)BATCH * P * DIM;
  const size_t idx = (size_t)blockIdx.x * 256 + threadIdx.x;
  if (idx >= tot) return;
  const float f = la[layer];
  const int c = (int)(idx % DIM);
  const size_t r = idx / DIM;
  const int p = (int)(r % P), b = (int)(r / P);
  const size_t po = ((size_t)b * P0 + p) * DIM + c;
  const float v = pos[po] * f;
  pos[po] = v;
  x[((size_t)(b * NP) + JTOK + p) * DIM + c] += v;
}

__global__ __launch_bounds__(256) void k_copy_in_x(const float* __restrict__ xin,
                                                   float* __restrict__ x) {
  const size_t tot = (size_t)BATCH * N0 * DIM;
  const size_t idx = (size_t)blockIdx.x * 256 + threadIdx.x;
  if (idx >= tot) return;
  const int c = (int)(idx % DIM);
  const size_t r = idx / DIM;
  const int m = (int)(r % N0), b = (int)(r / N0);
  x[((size_t)(b * NP) + m) * DIM + c] = xin[idx];
}

__global__ __launch_bounds__(256) void k_bcast_pos(const float* __restrict__ pin,
                                                   float* __restrict__ pos) {
  const size_t tot = (size_t)BATCH * P0 * DIM;
  const size_t idx = (size_t)blockIdx.x * 256 + threadIdx.x;
  if (idx >= tot) return;
  pos[idx] = pin[idx % ((size_t)P0 * DIM)];
}

__global__ __launch_bounds__(256) void k_copy_out_rows(const float* __restrict__ src,
                                                       float* __restrict__ dst,
                                                       int rows, int srcStride) {
  const size_t tot = (size_t)BATCH * rows * DIM;
  const size_t idx = (size_t)blockIdx.x * 256 + threadIdx.x;
  if (idx >= tot) return;
  const int c = (int)(idx % DIM);
  const size_t r = idx / DIM;
  const int m = (int)(r % rows), b = (int)(r / rows);
  dst[idx] = src[((size_t)b * srcStride + m) * DIM + c];
}

// ---------------------------------------------------------------------------
// Host orchestration
// ---------------------------------------------------------------------------
extern "C" void kernel_launch(void* const* d_in, const int* in_sizes, int n_in,
                              void* d_out, int out_size, void* d_ws, size_t ws_size,
                              hipStream_t stream) {
  const float* in_x    = (const float*)d_in[0];
  const float* in_fc   = (const float*)d_in[1];
  const float* in_la   = (const float*)d_in[2];
  const float* in_pos  = (const float*)d_in[3];
  const float* ln1_g   = (const float*)d_in[4];
  const float* ln1_b   = (const float*)d_in[5];
  const float* w_qkv   = (const float*)d_in[6];
  const float* w_out   = (const float*)d_in[7];
  const float* b_out   = (const float*)d_in[8];
  const float* ln2_g   = (const float*)d_in[9];
  const float* ln2_b   = (const float*)d_in[10];
  const float* w1      = (const float*)d_in[11];
  const float* b1      = (const float*)d_in[12];
  const float* w2      = (const float*)d_in[13];
  const float* b2      = (const float*)d_in[14];
  float* out = (float*)d_out;

  // workspace layout
  char* w = (char*)d_ws;
  auto take = [&](size_t bytes) { char* p = w; w += (bytes + 255) & ~(size_t)255; return p; };
  float*    xA      = (float*)take((size_t)BATCH * NP * DIM * 4);
  float*    xB      = (float*)take((size_t)BATCH * NP * DIM * 4);
  float*    posA    = (float*)take((size_t)BATCH * P0 * DIM * 4);
  float*    posB    = (float*)take((size_t)BATCH * P0 * DIM * 4);
  _Float16* hF16    = (_Float16*)take((size_t)BATCH * NP * DIM * 2);
  float*    qkvBuf  = (float*)take((size_t)BATCH * NP * 3 * DIM * 4);
  _Float16* qF      = (_Float16*)take((size_t)BATCH * HEADS * NP * HD * 2);
  _Float16* kF      = (_Float16*)take((size_t)BATCH * HEADS * NP * HD * 2);
  _Float16* vT      = (_Float16*)take((size_t)BATCH * HEADS * NP * HD * 2);
  float*    scBuf   = (float*)take((size_t)BATCH * HEADS * NP * NP * 4);
  _Float16* attn16  = (_Float16*)take((size_t)BATCH * HEADS * NP * NP * 2);
  _Float16* oF16    = (_Float16*)take((size_t)BATCH * NP * DIM * 2);
  _Float16* hidF16  = (_Float16*)take((size_t)BATCH * NP * MLP_ * 2);
  _Float16* wF16    = (_Float16*)take((size_t)MLP_ * DIM * 2);   // largest weight
  float*    human   = (float*)take((size_t)BATCH * P0 * 4);
  int*      topIdx  = (int*)take((size_t)BATCH * P0 * 4);

  // output offsets: x_out, attn[0..11], pos_out
  static const int nsAttn[DEPTH] = {273,273,273,273,197,197,197,143,143,143,106,106};
  size_t attn_off[DEPTH];
  size_t off = (size_t)BATCH * 106 * DIM;
  for (int l = 0; l < DEPTH; ++l) {
    attn_off[l] = off;
    off += (size_t)BATCH * HEADS * nsAttn[l] * nsAttn[l];
  }
  const size_t pos_off = off;

  auto cvt = [&](const float* src, size_t count) {
    k_cvt_f16<<<CEILDIV(count, (size_t)1024), 256, 0, stream>>>(src, wF16, count);
  };

  float* x = xA;   float* xAlt = xB;
  float* pos = posA; float* posAlt = posB;

  k_copy_in_x<<<CEILDIV((size_t)BATCH * N0 * DIM, 256), 256, 0, stream>>>(in_x, x);
  k_bcast_pos<<<CEILDIV((size_t)BATCH * P0 * DIM, 256), 256, 0, stream>>>(in_pos, pos);

  int n = N0, P = P0;
  for (int l = 0; l < DEPTH; ++l) {
    if (l > 0)
      k_pos_add<<<CEILDIV((size_t)BATCH * P * DIM, 256), 256, 0, stream>>>(
          pos, x, in_la, l, P);

    const int mtiles = CEILDIV(n, 16);

    // --- attention ---
    k_ln<<<dim3(n, BATCH), 256, 0, stream>>>(x, ln1_g + (size_t)l * DIM,
                                             ln1_b + (size_t)l * DIM, hF16);
    cvt(w_qkv + (size_t)l * 3 * DIM * DIM, (size_t)3 * DIM * DIM);
    k_gemm4<false, false, false, true, false><<<dim3(3 * DIM / 512, mtiles, BATCH),
        256, 0, stream>>>(hF16, wF16, nullptr, nullptr, qkvBuf, nullptr, DIM, 3 * DIM);
    k_rope<<<dim3(n, BATCH), 256, 0, stream>>>(
        qkvBuf, in_fc + (size_t)l * N0 * HD, qF, kF, vT);
    k_scores<<<dim3(CEILDIV(mtiles, 8), mtiles, BATCH * HEADS), 256, 0, stream>>>(
        qF, kF, scBuf, n);
    k_softmax<<<dim3(n, BATCH * HEADS), 256, 0, stream>>>(
        scBuf, out + attn_off[l], attn16, n);
    k_attnv<<<dim3(1, CEILDIV(mtiles, 2), BATCH * HEADS), 256, 0, stream>>>(
        attn16, vT, oF16, n);
    cvt(w_out + (size_t)l * DIM * DIM, (size_t)DIM * DIM);
    k_gemm4<true, false, true, true, false><<<dim3(DIM / 512, mtiles, BATCH),
        256, 0, stream>>>(oF16, wF16, b_out + (size_t)l * DIM, x, x, nullptr,
                          DIM, DIM);

    // --- pruning ---
    if (l == 3 || l == 6 || l == 9) {
      const int Pn = (P * 7 + 9) / 10;  // ceil(0.7*P): 180, 126, 89
      k_prune_scores<<<CEILDIV(BATCH * (n - JTOK), 256), 256, 0, stream>>>(
          out + attn_off[l], human, n);
      k_topk<<<BATCH, 256, 0, stream>>>(human, topIdx, n - JTOK, Pn);
      k_gather_x<<<CEILDIV((size_t)BATCH * (JTOK + Pn) * DIM, 256), 256, 0, stream>>>(
          x, xAlt, topIdx, Pn);
      k_gather_pos<<<CEILDIV((size_t)BATCH * Pn * DIM, 256), 256, 0, stream>>>(
          pos, posAlt, topIdx, Pn);
      float* t;
      t = x; x = xAlt; xAlt = t;
      t = pos; pos = posAlt; posAlt = t;
      P = Pn; n = JTOK + Pn;
    }

    // --- FFN ---
    const int mtF = CEILDIV(n, 16);
    k_ln<<<dim3(n, BATCH), 256, 0, stream>>>(x, ln2_g + (size_t)l * DIM,
                                             ln2_b + (size_t)l * DIM, hF16);
    cvt(w1 + (size_t)l * MLP_ * DIM, (size_t)MLP_ * DIM);
    k_gemm4<true, true, false, false, true><<<dim3(MLP_ / 512, mtF, BATCH),
        256, 0, stream>>>(hF16, wF16, b1 + (size_t)l * MLP_, nullptr, nullptr,
                          hidF16, DIM, MLP_);
    cvt(w2 + (size_t)l * DIM * MLP_, (size_t)DIM * MLP_);
    k_gemm4<true, false, true, true, false><<<dim3(DIM / 512, mtF, BATCH),
        256, 0, stream>>>(hidF16, wF16, b2 + (size_t)l * DIM, x, x, nullptr,
                          MLP_, DIM);
  }

  // final outputs: x (B,106,DIM) then pos (B,89,DIM)
  k_copy_out_rows<<<CEILDIV((size_t)BATCH * 106 * DIM, 256), 256, 0, stream>>>(
      x, out, 106, NP);
  k_copy_out_rows<<<CEILDIV((size_t)BATCH * 89 * DIM, 256), 256, 0, stream>>>(
      pos, out + pos_off, 89, P0);
}
